// TestRmsnormGroupFp8QuantModel_15530601742352
// MI455X (gfx1250) — 8-bit matrix operand order repaired and checked by reading, compiled, not run on hardware
//
#include <hip/hip_runtime.h>
#include <stdint.h>

#define H_DIM 4096
#define T_DIM 8192
#define KB_CNT (H_DIM / 128)
#define NT_CNT (H_DIM / 16)
#define FP8_MAX 448.0f

typedef int   v16i __attribute__((ext_vector_type(16)));
typedef float v8f  __attribute__((ext_vector_type(8)));

#if __has_builtin(__builtin_amdgcn_cvt_pk_fp8_f32)
__device__ inline unsigned pack_fp8x4(float a, float b, float c, float d) {
  int lo = __builtin_amdgcn_cvt_pk_fp8_f32(a, b, 0, false);
  int r  = __builtin_amdgcn_cvt_pk_fp8_f32(c, d, lo, true);
  return (unsigned)r;
}
#else
__device__ inline unsigned cvt1_fp8(float x) {
  unsigned s = (__float_as_uint(x) >> 24) & 0x80u;
  float ax = fabsf(x);
  if (!(ax == ax)) return 0x7Fu;
  if (ax >= 448.0f) return s | 0x7Eu;
  int e; float m = frexpf(ax, &e);
  int E = e + 6;
  if (E < 1) {
    int mi = (int)rintf(ax * 512.0f);
    return s | (unsigned)(mi > 7 ? 7 : mi);
  }
  int mi = (int)rintf(m * 16.0f);
  if (mi == 16) { mi = 8; ++E; }
  if (E > 15) return s | 0x7Eu;
  return s | (unsigned)((E << 3) | (mi & 7));
}
__device__ inline unsigned pack_fp8x4(float a, float b, float c, float d) {
  return cvt1_fp8(a) | (cvt1_fp8(b) << 8) | (cvt1_fp8(c) << 16) | (cvt1_fp8(d) << 24);
}
#endif

__device__ inline float wave_sum(float v) {
#pragma unroll
  for (int off = 16; off > 0; off >>= 1) v += __shfl_xor(v, off, 32);
  return v;
}

__global__ __launch_bounds__(256) void k_prep(
    const float* __restrict__ xin, float* __restrict__ resid,
    const float* __restrict__ nw, uint8_t* __restrict__ qperm,
    float* __restrict__ as_t, int do_relu)
{
  const int m    = blockIdx.x;
  const int t    = threadIdx.x;
  const int lane = t & 31, wid = t >> 5;
  const float* src = do_relu ? (xin + (size_t)m * H_DIM)
                             : (resid + (size_t)m * H_DIM);
  float y[16];
  float ss = 0.0f;
#pragma unroll
  for (int c = 0; c < 4; ++c) {
    float4 v = *(const float4*)(src + t * 16 + c * 4);
    if (do_relu) {
      v.x = fmaxf(v.x, 0.0f); v.y = fmaxf(v.y, 0.0f);
      v.z = fmaxf(v.z, 0.0f); v.w = fmaxf(v.w, 0.0f);
    }
    y[c*4+0] = v.x; y[c*4+1] = v.y; y[c*4+2] = v.z; y[c*4+3] = v.w;
    ss += v.x*v.x + v.y*v.y + v.z*v.z + v.w*v.w;
  }
  if (do_relu) {
#pragma unroll
    for (int c = 0; c < 4; ++c) {
      float4 v = { y[c*4+0], y[c*4+1], y[c*4+2], y[c*4+3] };
      *(float4*)(resid + (size_t)m * H_DIM + t * 16 + c * 4) = v;
    }
  }
  __shared__ float sred[8];
  float s = wave_sum(ss);
  if (lane == 0) sred[wid] = s;
  __syncthreads();
  if (t == 0) {
    float tot = 0.0f;
#pragma unroll
    for (int i = 0; i < 8; ++i) tot += sred[i];
    sred[0] = tot;
  }
  __syncthreads();
  const float rrms = rsqrtf(sred[0] * (1.0f / H_DIM) + 1e-6f);

  float amax = 0.0f;
#pragma unroll
  for (int c = 0; c < 4; ++c) {
    float4 w = *(const float4*)(nw + t * 16 + c * 4);
    y[c*4+0] *= rrms * w.x; y[c*4+1] *= rrms * w.y;
    y[c*4+2] *= rrms * w.z; y[c*4+3] *= rrms * w.w;
  }
#pragma unroll
  for (int i = 0; i < 16; ++i) amax = fmaxf(amax, fabsf(y[i]));
#pragma unroll
  for (int off = 1; off < 8; off <<= 1) amax = fmaxf(amax, __shfl_xor(amax, off, 32));

  const float am    = fmaxf(amax, 1e-4f);
  const float scale = am * (1.0f / FP8_MAX);
  const float inv   = FP8_MAX / am;
  const int   kb    = t >> 3;
  if ((t & 7) == 0) as_t[(size_t)kb * T_DIM + m] = scale;

  const int mt = m >> 4, ml = m & 15, chunk = t & 7;
  uint32_t* qbase = (uint32_t*)(qperm + ((size_t)(mt * KB_CNT + kb) << 11));
#pragma unroll
  for (int c = 0; c < 4; ++c) {
    unsigned d = pack_fp8x4(y[c*4+0]*inv, y[c*4+1]*inv, y[c*4+2]*inv, y[c*4+3]*inv);
    const int dlane = ml + ((c >= 2) ? 16 : 0);
    const int v     = chunk * 2 + (c & 1);
    qbase[dlane * 16 + v] = d;
  }
}

__global__ __launch_bounds__(256) void k_wconv(
    const float* __restrict__ w, uint8_t* __restrict__ wq)
{
  const int g     = blockIdx.x * 256 + threadIdx.x;
  const int brick = g >> 9;
  const int r     = g & 511;
  const int lane  = r >> 4;
  const int v     = r & 15;
  const int kb    = brick >> 8;
  const int nt    = brick & (NT_CNT - 1);
  const int n     = nt * 16 + (lane & 15);
  const int kbase = kb * 128 + (v >> 1) * 16 + ((lane >= 16) ? 8 : 0) + (v & 1) * 4;
  const float* p  = w + (size_t)kbase * H_DIM + n;
  ((uint32_t*)wq)[g] = pack_fp8x4(p[0], p[H_DIM], p[2 * H_DIM], p[3 * H_DIM]);
}

__global__ __launch_bounds__(256) void k_gemm(
    const uint8_t* __restrict__ qperm, const float* __restrict__ as_t,
    const uint8_t* __restrict__ wq,    const float* __restrict__ ws,
    float* __restrict__ resid)
{
  const int lane = threadIdx.x & 31;
  const int wid  = threadIdx.x >> 5;
  const int mw   = wid & 3, nw = wid >> 2;
  const int mt0  = blockIdx.y * 8 + mw * 2;
  const int nt0  = blockIdx.x * 8 + nw * 4;
  const int nblk = blockIdx.x;
  const int hl   = lane >> 4;

  v8f acc[2][4];
#pragma unroll
  for (int i = 0; i < 2; ++i)
#pragma unroll
    for (int j = 0; j < 4; ++j) acc[i][j] = (v8f){0,0,0,0,0,0,0,0};

  for (int kb = 0; kb < KB_CNT; ++kb) {
    v16i a[2], b[4];
#pragma unroll
    for (int i = 0; i < 2; ++i)
      a[i] = *(const v16i*)(qperm + ((size_t)((mt0 + i) * KB_CNT + kb) << 11) + (lane << 6));
#pragma unroll
    for (int j = 0; j < 4; ++j)
      b[j] = *(const v16i*)(wq + ((size_t)(kb * NT_CNT + nt0 + j) << 11) + (lane << 6));

    if (kb + 1 < KB_CNT) {
      __builtin_prefetch(qperm + ((size_t)(mt0 * KB_CNT + kb + 1) << 11) + (lane << 6), 0, 1);
      __builtin_prefetch(wq + ((size_t)((kb + 1) * NT_CNT + nt0) << 11) + (lane << 6), 0, 1);
    }

    const float ws_kb = ws[kb * (H_DIM / 128) + nblk];
    float sa[2][8];
#pragma unroll
    for (int i = 0; i < 2; ++i) {
      const float4 lo = *(const float4*)(as_t + (size_t)kb * T_DIM + (mt0 + i) * 16 + hl * 8);
      const float4 hi = *(const float4*)(as_t + (size_t)kb * T_DIM + (mt0 + i) * 16 + hl * 8 + 4);
      sa[i][0] = lo.x * ws_kb; sa[i][1] = lo.y * ws_kb;
      sa[i][2] = lo.z * ws_kb; sa[i][3] = lo.w * ws_kb;
      sa[i][4] = hi.x * ws_kb; sa[i][5] = hi.y * ws_kb;
      sa[i][6] = hi.z * ws_kb; sa[i][7] = hi.w * ws_kb;
    }

#pragma unroll
    for (int i = 0; i < 2; ++i)
#pragma unroll
      for (int j = 0; j < 4; ++j) {
        v8f zero = (v8f){0,0,0,0,0,0,0,0};
        v8f d = __builtin_amdgcn_wmma_f32_16x16x128_fp8_fp8(
            a[i], b[j], (short)0, zero, false, false);
#pragma unroll
        for (int v = 0; v < 8; ++v) acc[i][j][v] += d[v] * sa[i][v];
      }
  }

#pragma unroll
  for (int i = 0; i < 2; ++i)
#pragma unroll
    for (int j = 0; j < 4; ++j)
#pragma unroll
      for (int v = 0; v < 8; ++v) {
        const int row = (mt0 + i) * 16 + v + hl * 8;
        const int col = (nt0 + j) * 16 + (lane & 15);
        float* p = resid + (size_t)row * H_DIM + col;
        *p = *p + acc[i][j][v];
      }
}

__global__ __launch_bounds__(256) void k_final(
    float* __restrict__ resid, const float* __restrict__ nw)
{
  const int m = blockIdx.x;
  const int t = threadIdx.x;
  const int lane = t & 31, wid = t >> 5;
  float y[16];
  float ss = 0.0f;
#pragma unroll
  for (int c = 0; c < 4; ++c) {
    float4 v = *(const float4*)(resid + (size_t)m * H_DIM + t * 16 + c * 4);
    y[c*4+0] = v.x; y[c*4+1] = v.y; y[c*4+2] = v.z; y[c*4+3] = v.w;
    ss += v.x*v.x + v.y*v.y + v.z*v.z + v.w*v.w;
  }
  __shared__ float sred[8];
  float s = wave_sum(ss);
  if (lane == 0) sred[wid] = s;
  __syncthreads();
  if (t == 0) {
    float tot = 0.0f;
#pragma unroll
    for (int i = 0; i < 8; ++i) tot += sred[i];
    sred[0] = tot;
  }
  __syncthreads();
  const float rrms = rsqrtf(sred[0] * (1.0f / H_DIM) + 1e-6f);
#pragma unroll
  for (int c = 0; c < 4; ++c) {
    float4 w = *(const float4*)(nw + t * 16 + c * 4);
    float4 o = { y[c*4+0]*rrms*w.x, y[c*4+1]*rrms*w.y,
                 y[c*4+2]*rrms*w.z, y[c*4+3]*rrms*w.w };
    *(float4*)(resid + (size_t)m * H_DIM + t * 16 + c * 4) = o;
  }
}

extern "C" void kernel_launch(void* const* d_in, const int* in_sizes, int n_in,
                              void* d_out, int out_size, void* d_ws, size_t ws_size,
                              hipStream_t stream) {
  const float* x   = (const float*)d_in[0];
  const float* w[3]  = { (const float*)d_in[1], (const float*)d_in[2], (const float*)d_in[3] };
  const float* wsc[3]= { (const float*)d_in[4], (const float*)d_in[5], (const float*)d_in[6] };
  const float* nw[4] = { (const float*)d_in[7], (const float*)d_in[8],
                         (const float*)d_in[9], (const float*)d_in[10] };
  float* resid = (float*)d_out;

  uint8_t* wsb   = (uint8_t*)d_ws;
  uint8_t* qperm = wsb;
  float*   as_t  = (float*)(wsb + (size_t)T_DIM * H_DIM);
  uint8_t* wq    = wsb + (size_t)T_DIM * H_DIM
                       + (size_t)KB_CNT * T_DIM * sizeof(float);

  const dim3 blk(256);
  const dim3 gPrep(T_DIM);
  const dim3 gConv((H_DIM / 4) * (H_DIM / 256));
  const dim3 gGemm(H_DIM / 128, T_DIM / 128);

  for (int layer = 0; layer < 3; ++layer) {
    k_prep<<<gPrep, blk, 0, stream>>>(x, resid, nw[layer], qperm, as_t,
                                      layer == 0 ? 1 : 0);
    k_wconv<<<gConv, blk, 0, stream>>>(w[layer], wq);
    k_gemm<<<gGemm, blk, 0, stream>>>(qperm, as_t, wq, wsc[layer], resid);
  }
  k_final<<<gPrep, blk, 0, stream>>>(resid, nw[3]);
}
